// AttentivePooling_20031727469162
// MI455X (gfx1250) — matrix operand order repaired and checked by reading, not compiled, not run on hardware
//
#include <hip/hip_runtime.h>
#include <hip/hip_bf16.h>

// AttentivePooling for MI455X (gfx1250, wave32, WMMA bf16).
// B=16, C=5120, T=1024, Cr=256.
//   GEMM1: h_t[b][t][r] = tanh(BN(relu(W1a @ x[b] + const1)))   (bf16, t-major)
//   GEMM2: logits = W2 @ h, softmax over t + attentive mu/sigma, fused.
// 86 GFLOP via v_wmma_f32_16x16x32_bf16; HBM traffic ~1 GB (3 passes over x),
// all weight/h operands are L2-resident (192 MB L2).

#define B_   16
#define C_   5120
#define T_   1024
#define CR_  256
#define CIN_ (3 * C_)

typedef __attribute__((ext_vector_type(16))) __bf16 v16bf;
typedef __attribute__((ext_vector_type(2)))  __bf16 v2bf;
typedef __attribute__((ext_vector_type(8)))  float  v8f;

// native bf16 conversion (backend lowers fptrunc -> hw cvt, RTNE)
__device__ __forceinline__ unsigned short f2bf(float f) {
  union { __bf16 h; unsigned short u; } cv;
  cv.h = (__bf16)f;
  return cv.u;
}
// packed pair: build_vector of two fptruncs -> single v_cvt_pk_bf16_f32
__device__ __forceinline__ unsigned int f2bf2(float lo, float hi) {
  union { v2bf v; unsigned int u; } cv;
  cv.v[0] = (__bf16)lo;
  cv.v[1] = (__bf16)hi;
  return cv.u;
}

__device__ __forceinline__ v16bf mk16(uint4 lo, uint4 hi) {
  union { uint4 u[2]; v16bf v; } w;
  w.u[0] = lo; w.u[1] = hi;
  return w.v;
}

__device__ __forceinline__ float wred_sum(float v) {
#pragma unroll
  for (int m = 16; m >= 1; m >>= 1) v += __shfl_xor(v, m, 32);
  return v;
}
__device__ __forceinline__ float wred_max(float v) {
#pragma unroll
  for (int m = 16; m >= 1; m >>= 1) v = fmaxf(v, __shfl_xor(v, m, 32));
  return v;
}

// ---------------------------------------------------------------------------
// Kernel 1: per-(b,c) mean and unbiased std over T. One wave per row.
// ---------------------------------------------------------------------------
__global__ __launch_bounds__(256) void k_stats(const float* __restrict__ x,
                                               float* __restrict__ gmean,
                                               float* __restrict__ gstd) {
  const int wave = threadIdx.x >> 5, lane = threadIdx.x & 31;
  const size_t row = (size_t)blockIdx.x * 8 + wave;      // 81920 rows
  const float4* p = reinterpret_cast<const float4*>(x + row * T_);
  float4 v[8];
#pragma unroll
  for (int i = 0; i < 8; ++i) v[i] = p[lane + i * 32];   // 8 b128 loads in flight
  float s = 0.f, ss = 0.f;
#pragma unroll
  for (int i = 0; i < 8; ++i) {
    s  += v[i].x + v[i].y + v[i].z + v[i].w;
    ss += v[i].x * v[i].x + v[i].y * v[i].y + v[i].z * v[i].z + v[i].w * v[i].w;
  }
  s = wred_sum(s); ss = wred_sum(ss);
  if (lane == 0) {
    float mean = s * (1.0f / T_);
    float var  = (ss - (float)T_ * mean * mean) * (1.0f / (T_ - 1));
    gmean[row] = mean;
    gstd[row]  = sqrtf(fmaxf(var, 0.f));
  }
}

// ---------------------------------------------------------------------------
// Kernel 2: convert w1[:, :C] and w2 to bf16 once (L2-resident afterwards).
// ---------------------------------------------------------------------------
__global__ __launch_bounds__(256) void k_convert(const float* __restrict__ w1,
                                                 const float* __restrict__ w2,
                                                 unsigned short* __restrict__ w1b,
                                                 unsigned short* __restrict__ w2b) {
  const size_t i = (size_t)blockIdx.x * blockDim.x + threadIdx.x;  // 1,310,720
  const size_t o = i / C_, c = i - o * C_;
  w1b[i] = f2bf(w1[o * CIN_ + c]);
  w2b[i] = f2bf(w2[i]);
}

// ---------------------------------------------------------------------------
// Kernel 3: fold the (mean,std) branch of w1 and b1 into const1[b][o].
// ---------------------------------------------------------------------------
__global__ __launch_bounds__(256) void k_ctx(const float* __restrict__ w1,
                                             const float* __restrict__ b1,
                                             const float* __restrict__ gmean,
                                             const float* __restrict__ gstd,
                                             float* __restrict__ const1) {
  __shared__ float sd[B_][256];
  const int o = blockIdx.x, tid = threadIdx.x;
  float acc[B_];
#pragma unroll
  for (int b = 0; b < B_; ++b) acc[b] = 0.f;
  const float* wm = w1 + (size_t)o * CIN_ + C_;
  const float* ws = w1 + (size_t)o * CIN_ + 2 * C_;
  for (int k = tid; k < C_; k += 256) {
    float a = wm[k], c = ws[k];
#pragma unroll
    for (int b = 0; b < B_; ++b)
      acc[b] += a * gmean[b * C_ + k] + c * gstd[b * C_ + k];
  }
#pragma unroll
  for (int b = 0; b < B_; ++b) sd[b][tid] = acc[b];
  __syncthreads();
  for (int s = 128; s > 0; s >>= 1) {
    if (tid < s) {
#pragma unroll
      for (int b = 0; b < B_; ++b) sd[b][tid] += sd[b][tid + s];
    }
    __syncthreads();
  }
  if (tid < B_) const1[tid * CR_ + o] = sd[tid][0] + b1[o];
}

// ---------------------------------------------------------------------------
// Kernel 4: h = tanh(BN(relu(W1a @ x[b] + const1))) written TRANSPOSED as
// h_t[b][t][r] (bf16) so GEMM2 can fragment-load it straight from L2.
// WG tile 128(M) x 128(N), 8 waves (4M x 2N), wave tile 32x64.
// A frags: direct global b128 from L2-resident w1b.
// B tile: x f32 -> bf16, double-buffered LDS [n][k] (rows padded to 40 halfs,
// lane banks l*20 mod 64 all distinct => conflict-minimal b128 frag loads).
// Epilogue: LDS transpose tile -> fully coalesced b128 global stores.
// ---------------------------------------------------------------------------
__global__ __launch_bounds__(256) void k_gemm1(
    const unsigned short* __restrict__ w1b, const float* __restrict__ x,
    const float* __restrict__ const1,
    const float* __restrict__ gamma, const float* __restrict__ beta,
    const float* __restrict__ rmean, const float* __restrict__ rvar,
    unsigned short* __restrict__ hbt) {
  // region reused: double-buffered sB (2*128*40 halfs = 20480B) then
  // transpose tile sT (128*136 halfs = 34816B)
  __shared__ __align__(16) unsigned short smem[128 * 136];
  unsigned short (*sB)[128][40] =
      reinterpret_cast<unsigned short (*)[128][40]>(smem);
  unsigned short (*sT)[136] = reinterpret_cast<unsigned short (*)[136]>(smem);
  __shared__ float sC[128], sScale[128], sShift[128];

  const int b = blockIdx.z;
  const int m0 = blockIdx.y * 128;
  const int n0 = blockIdx.x * 128;
  const int tid = threadIdx.x, lane = tid & 31;
  const int wv = tid >> 5;
  const int wm = wv >> 1;          // m offset 32*wm
  const int wn = wv & 1;           // n offset 64*wn
  const int hh = lane >> 4;

  if (tid < 128) {
    int o = m0 + tid;
    float inv = gamma[o] * rsqrtf(rvar[o] + 1e-5f);
    sScale[tid] = inv;
    sShift[tid] = beta[o] - rmean[o] * inv;
    sC[tid]     = const1[b * CR_ + o];
  }

  v8f acc[2][4];
  const v8f z8 = {0.f, 0.f, 0.f, 0.f, 0.f, 0.f, 0.f, 0.f};
#pragma unroll
  for (int i = 0; i < 2; ++i)
#pragma unroll
    for (int j = 0; j < 4; ++j) acc[i][j] = z8;

  const float* xb = x + (size_t)b * C_ * T_;
  const uint4* w1v = reinterpret_cast<const uint4*>(w1b);

  // staging register buffers: i in {0,1}: g = i*256+tid, kp = g>>5 (k pair),
  // n4 = (g&31)*4. Per wave kp is uniform, lanes span n -> coalesced b128.
  int kp[2], n4[2];
#pragma unroll
  for (int i = 0; i < 2; ++i) {
    int g = i * 256 + tid;
    kp[i] = g >> 5;
    n4[i] = (g & 31) * 4;
  }
  float4 ra[2], rb[2];

#define LOADX(KK)                                                            \
  _Pragma("unroll") for (int i = 0; i < 2; ++i) {                            \
    const float* gp = &xb[(size_t)((KK) + 2 * kp[i]) * T_ + n0 + n4[i]];     \
    ra[i] = *reinterpret_cast<const float4*>(gp);                            \
    rb[i] = *reinterpret_cast<const float4*>(gp + T_);                       \
  }

  LOADX(0);
  int buf = 0;
  for (int kk = 0; kk < C_; kk += 32) {
    // pack k-pairs (one v_cvt_pk_bf16_f32 each) -> b32 LDS stores
#pragma unroll
    for (int i = 0; i < 2; ++i) {
      const float* fa = &ra[i].x;
      const float* fb = &rb[i].x;
#pragma unroll
      for (int j = 0; j < 4; ++j) {
        *reinterpret_cast<unsigned int*>(&sB[buf][n4[i] + j][2 * kp[i]]) =
            f2bf2(fa[j], fb[j]);
      }
    }
    __syncthreads();
    if (kk + 32 < C_) { LOADX(kk + 32); }   // overlap next loads with WMMAs

#pragma unroll
    for (int ms = 0; ms < 2; ++ms) {
      int row = m0 + wm * 32 + ms * 16 + (lane & 15);
      int aidx = row * 640 + (kk >> 3) + hh;          // 5120 halfs = 640 uint4
      v16bf af = mk16(w1v[aidx], w1v[aidx + 2]);
#pragma unroll
      for (int ns = 0; ns < 4; ++ns) {
        const uint4* rp = reinterpret_cast<const uint4*>(
            &sB[buf][wn * 64 + ns * 16 + (lane & 15)][0]);
        v16bf bf = mk16(rp[hh], rp[hh + 2]);
        acc[ms][ns] = __builtin_amdgcn_wmma_f32_16x16x32_bf16(
            false, af, false, bf, (short)0, acc[ms][ns], false, false);
      }
    }
    buf ^= 1;
  }
#undef LOADX
  __syncthreads();   // done with sB region; reuse as transpose tile

  // epilogue math -> LDS transpose tile [t_local][m_local]
#pragma unroll
  for (int ms = 0; ms < 2; ++ms)
#pragma unroll
    for (int ns = 0; ns < 4; ++ns)
#pragma unroll
      for (int r = 0; r < 8; ++r) {
        int rl = wm * 32 + ms * 16 + r + 8 * hh;
        int tl = wn * 64 + ns * 16 + (lane & 15);
        float v = acc[ms][ns][r] + sC[rl];
        v = fmaxf(v, 0.f);
        v = v * sScale[rl] + sShift[rl];
        v = tanhf(v);
        sT[tl][rl] = f2bf(v);
      }
  __syncthreads();

  // coalesced b128 writeout: h_t[b][n0+t][m0 .. m0+127]
  uint4* hv = reinterpret_cast<uint4*>(hbt);
#pragma unroll
  for (int i = 0; i < 8; ++i) {
    int e = i * 256 + tid;
    int t = e >> 4, c4 = e & 15;
    uint4 val = reinterpret_cast<const uint4*>(&sT[t][0])[c4];
    hv[(size_t)(b * T_ + n0 + t) * 32 + (m0 >> 3) + c4] = val;
  }
}

// ---------------------------------------------------------------------------
// Kernel 5: logits = W2 @ h (16 channels x 1024 t per WG, K=256), softmax
// over t + attentive mean/std vs x. A and B fragments are direct global
// b128 loads from L2-resident w2b / h_t: NO staging LDS, NO barriers in the
// GEMM phase. LDS (64 KB dynamic) holds only the logits tile.
// ---------------------------------------------------------------------------
__global__ __launch_bounds__(256) void k_attn(const unsigned short* __restrict__ w2b,
                                              const unsigned short* __restrict__ hbt,
                                              const float* __restrict__ x,
                                              float* __restrict__ out) {
  extern __shared__ __align__(16) unsigned char dynsm[];
  float* slog = reinterpret_cast<float*>(dynsm);     // [16][1024]

  const int c0 = blockIdx.x * 16;
  const int b  = blockIdx.y;
  const int tid = threadIdx.x, lane = tid & 31, wv = tid >> 5;
  const int hh = lane >> 4;

  v8f acc[8];
  const v8f z8 = {0.f, 0.f, 0.f, 0.f, 0.f, 0.f, 0.f, 0.f};
#pragma unroll
  for (int i = 0; i < 8; ++i) acc[i] = z8;

  const uint4* w2v = reinterpret_cast<const uint4*>(w2b);
  const uint4* hv  = reinterpret_cast<const uint4*>(hbt) +
                     (size_t)b * T_ * (CR_ / 8);

#pragma unroll
  for (int kk = 0; kk < CR_; kk += 32) {
    int row = c0 + (lane & 15);
    int aidx = row * 32 + (kk >> 3) + hh;            // 256 halfs = 32 uint4/row
    v16bf af = mk16(w2v[aidx], w2v[aidx + 2]);
#pragma unroll
    for (int ns = 0; ns < 8; ++ns) {
      int t = wv * 128 + ns * 16 + (lane & 15);
      int bidx = t * 32 + (kk >> 3) + hh;
      v16bf bf = mk16(hv[bidx], hv[bidx + 2]);
      acc[ns] = __builtin_amdgcn_wmma_f32_16x16x32_bf16(
          false, af, false, bf, (short)0, acc[ns], false, false);
    }
  }

  // spill logits to LDS (b2 omitted: softmax is shift-invariant per row)
#pragma unroll
  for (int ns = 0; ns < 8; ++ns)
#pragma unroll
    for (int r = 0; r < 8; ++r) {
      int m = r + 8 * hh;
      int t = wv * 128 + ns * 16 + (lane & 15);
      slog[m * T_ + t] = acc[ns][r];
    }
  __syncthreads();

  // each wave: softmax + weighted stats for 2 channel rows
#pragma unroll
  for (int rr = 0; rr < 2; ++rr) {
    int m = wv * 2 + rr;
    const float* lrow = slog + m * T_;
    float lm = -3.4e38f;
#pragma unroll
    for (int i = 0; i < 32; ++i) lm = fmaxf(lm, lrow[lane + 32 * i]);
    lm = wred_max(lm);

    const float* xrow = x + ((size_t)b * C_ + c0 + m) * T_;
    float s = 0.f, mn = 0.f, e2 = 0.f;
#pragma unroll
    for (int i = 0; i < 32; ++i) {
      int t = lane + 32 * i;
      float e  = __expf(lrow[t] - lm);
      float xv = xrow[t];
      s += e; mn += e * xv; e2 += e * xv * xv;
    }
    s = wred_sum(s); mn = wred_sum(mn); e2 = wred_sum(e2);
    if (lane == 0) {
      float inv = 1.0f / s;
      float mu  = mn * inv;
      float var = e2 * inv - mu * mu;
      out[(size_t)b * (2 * C_) + c0 + m]      = mu;
      out[(size_t)b * (2 * C_) + C_ + c0 + m] = sqrtf(fmaxf(var, 1e-5f));
    }
  }
}

// ---------------------------------------------------------------------------
extern "C" void kernel_launch(void* const* d_in, const int* in_sizes, int n_in,
                              void* d_out, int out_size, void* d_ws, size_t ws_size,
                              hipStream_t stream) {
  const float* x     = (const float*)d_in[0];
  const float* w1    = (const float*)d_in[1];
  const float* b1    = (const float*)d_in[2];
  const float* gamma = (const float*)d_in[3];
  const float* beta  = (const float*)d_in[4];
  const float* rmean = (const float*)d_in[5];
  const float* rvar  = (const float*)d_in[6];
  const float* w2    = (const float*)d_in[7];
  // d_in[8] (b2) intentionally unused: softmax over T is shift-invariant.
  float* out = (float*)d_out;

  char* ws = (char*)d_ws;
  float* gmean          = (float*)(ws);                       // 16*5120 f32
  float* gstd           = (float*)(ws + 327680);              // 16*5120 f32
  float* const1         = (float*)(ws + 655360);              // 16*256  f32
  unsigned short* w1b   = (unsigned short*)(ws + 671744);     // 256*5120 bf16
  unsigned short* w2b   = (unsigned short*)(ws + 3293184);    // 5120*256 bf16
  unsigned short* hbt   = (unsigned short*)(ws + 5914624);    // 16*1024*256 bf16 (t-major)

  k_stats  <<<10240, 256, 0, stream>>>(x, gmean, gstd);
  k_convert<<<5120, 256, 0, stream>>>(w1, w2, w1b, w2b);
  k_ctx    <<<256, 256, 0, stream>>>(w1, b1, gmean, gstd, const1);
  k_gemm1  <<<dim3(8, 2, 16), 256, 0, stream>>>(w1b, x, const1, gamma, beta,
                                                rmean, rvar, hbt);
  k_attn   <<<dim3(320, 16), 256, 65536, stream>>>(w2b, hbt, x, out);

  (void)in_sizes; (void)n_in; (void)out_size; (void)ws_size;
}
